// CGPT_59725815218914
// MI455X (gfx1250) — compile-verified, matrix operand order checked by reading, not run on hardware
//
#include <hip/hip_runtime.h>
#include <hip/hip_bf16.h>
#include <math.h>

// ---------------- model constants (from reference) ----------------
#define HDIM    256
#define NHEAD   8
#define HD      32
#define NINNER  1024
#define BATCH   4
#define T1      16384
#define T2      2048

typedef __attribute__((ext_vector_type(16))) __bf16 v16bf;
typedef __attribute__((ext_vector_type(8)))  __bf16 v8bf;
typedef __attribute__((ext_vector_type(8)))  float  v8f;
typedef __attribute__((ext_vector_type(4)))  unsigned v4u;
typedef __attribute__((ext_vector_type(8)))  int      v8i;
typedef __attribute__((ext_vector_type(4)))  int      v4i;

#if __has_builtin(__builtin_amdgcn_tensor_load_to_lds) && __has_builtin(__builtin_amdgcn_s_wait_tensorcnt)
#define ATHENA_TDM 1
#else
#define ATHENA_TDM 0
#endif

__device__ __forceinline__ float gelu_f(float x) {
    return 0.5f * x * (1.0f + erff(x * 0.70710678118654752f));
}

// ---------------- f32 -> bf16 weight cast (RNE) ----------------
__global__ __launch_bounds__(256) void k_cast_bf16(
    const float* __restrict__ in, unsigned short* __restrict__ out, int n)
{
    int i = blockIdx.x * 256 + threadIdx.x;
    if (i < n) {
        unsigned u = __builtin_bit_cast(unsigned, in[i]);
        out[i] = (unsigned short)((u + 0x7FFFu + ((u >> 16) & 1u)) >> 16);
    }
}

// ---------------- zero fill ----------------
__global__ __launch_bounds__(256) void k_zero(float* __restrict__ p, int n)
{
    int i = blockIdx.x * 256 + threadIdx.x;
    if (i < n) p[i] = 0.f;
}

// ---------------- WMMA GEMM: C[M,N] = epilogue(A[M,K] * W[K,N] + bias) ----------------
// mode 0: +bias    1: gelu(+bias)    2: gelu(+bias)+R    3: (+bias)+R
// FAST: K%32==0 && N%128==0 && M%128==0 (no edge guards)
// 128 threads = 4 waves; each wave owns a 64x64 C-tile (4x4 WMMA fragments).
// Double-buffered pipeline: TDM (A tile) + W global loads for tile i+1 are in
// flight while tile i runs through 16 WMMAs; one barrier per K-step.
template <bool FAST>
__global__ __launch_bounds__(128) void k_gemm_wmma(
    const float* __restrict__ A, const unsigned short* __restrict__ Wq,
    const float* __restrict__ bias, const float* __restrict__ R,
    float* __restrict__ C, int M, int N, int K, int mode)
{
    constexpr int BM = 128, BN = 128, BK = 32;
    constexpr int LDA = BK + 8;   // f32 elems; 160B rows; matches TDM pad 32dw+8dw
    constexpr int LDB = BK + 8;   // bf16 elems; 80B rows
    __shared__ float  sAf[2][BM * LDA];   // A tiles (f32, TDM destination), 2x20KB
    __shared__ __bf16 sB[2][BN * LDB];    // W tiles transposed sB[.][n*LDB+k], 2x10KB

    const int tid  = threadIdx.x;        // 0..127
    const int bm   = blockIdx.x * BM;
    const int bn   = blockIdx.y * BN;
    const int wave = tid >> 5;           // 0..3
    const int lane = tid & 31;
    const int wm   = (wave >> 1) * 64;   // 2 waves along M
    const int wn   = (wave & 1) * 64;    // 2 waves along N
    const int lrow = lane & 15;
    const int kh   = (lane >> 4) * 8;    // K-half select per ISA 16-bit A layout

    unsigned short brg[32];   // W-tile staging registers (32 per thread)
#if !ATHENA_TDM
    float4 arg4[8];           // A-tile staging registers (manual fallback)
#endif

    auto load_b_regs = [&](int k0) {
        for (int g = 0; g < 32; ++g) {
            int idx = tid + g * 128;
            int kk = idx >> 7, nn = idx & 127;
            if (FAST) {
                brg[g] = Wq[(size_t)(k0 + kk) * N + bn + nn];
            } else {
                unsigned short u = 0;
                if (k0 + kk < K && bn + nn < N) u = Wq[(size_t)(k0 + kk) * N + bn + nn];
                brg[g] = u;
            }
        }
    };
    auto write_b = [&](int buf) {
        for (int g = 0; g < 32; ++g) {
            int idx = tid + g * 128;
            int kk = idx >> 7, nn = idx & 127;
            sB[buf][nn * LDB + kk] = __builtin_bit_cast(__bf16, brg[g]);
        }
    };

#if ATHENA_TDM
    // ---- Tensor Data Mover: 2D tile (BM rows x BK f32) global -> LDS ----
    // OOB rows/cols return zero via tensor_dim clamping (handles K=3 etc.);
    // LDS row padding 32dw + 8dw via pad_interval=4 (32 DW), pad_amount=7 (8 DW).
    auto issue_a = [&](int k0, int buf) {
        if (tid < 32) {
            unsigned long long ga =
                (unsigned long long)(uintptr_t)(A + (size_t)bm * K + k0);
            unsigned lds = (unsigned)(uintptr_t)(&sAf[buf][0]);
            unsigned td0 = (unsigned)(K - k0);   // remaining cols (f32 units)
            unsigned td1 = (unsigned)(M - bm);   // remaining rows
            v4u g0;
            g0[0] = 1u;                           // count=1, no gather
            g0[1] = lds;                          // lds_addr
            g0[2] = (unsigned)ga;                 // global_addr[31:0]
            g0[3] = (unsigned)((ga >> 32) & 0x1FFFFFFu) | 0x80000000u; // addr[56:32] | type=2
            v8i g1;
            g1[0] = (int)((2u << 16) | (1u << 20) | (4u << 22) | (7u << 25)); // 4B data, pad on
            g1[1] = (int)((td0 & 0xFFFFu) << 16);                              // dim0[15:0]
            g1[2] = (int)(((td0 >> 16) & 0xFFFFu) | ((td1 & 0xFFFFu) << 16));  // dim0 hi | dim1 lo
            g1[3] = (int)(((td1 >> 16) & 0xFFFFu) | ((unsigned)BK << 16));     // dim1 hi | tile_dim0
            g1[4] = BM;                                                        // tile_dim1
            g1[5] = (int)K;                                                    // dim0_stride lo
            g1[6] = 0;
            g1[7] = 0;
            v4i z4 = {0, 0, 0, 0};
#if __clang_major__ >= 23
            v8i z8 = {0, 0, 0, 0, 0, 0, 0, 0};
            __builtin_amdgcn_tensor_load_to_lds(g0, g1, z4, z4, z8, 0);
#else
            __builtin_amdgcn_tensor_load_to_lds(g0, g1, z4, z4, 0);
#endif
        }
    };
    auto wait_a = [&]() {
        if (tid < 32) __builtin_amdgcn_s_wait_tensorcnt(0);
    };
#else
    auto load_a_regs = [&](int k0) {
        for (int g = 0; g < 8; ++g) {
            int idx4 = tid + g * 128;
            int r = idx4 >> 3;
            int c = (idx4 & 7) << 2;
            if (FAST) {
                arg4[g] = *(const float4*)(A + (size_t)(bm + r) * K + k0 + c);
            } else {
                float x0 = 0.f, x1 = 0.f, x2 = 0.f, x3 = 0.f;
                int grow = bm + r;
                if (grow < M) {
                    const float* ap = A + (size_t)grow * K + k0 + c;
                    if (k0 + c + 4 <= K) {
                        float4 v4 = *(const float4*)ap;
                        x0 = v4.x; x1 = v4.y; x2 = v4.z; x3 = v4.w;
                    } else {
                        if (k0 + c + 0 < K) x0 = ap[0];
                        if (k0 + c + 1 < K) x1 = ap[1];
                        if (k0 + c + 2 < K) x2 = ap[2];
                    }
                }
                arg4[g].x = x0; arg4[g].y = x1; arg4[g].z = x2; arg4[g].w = x3;
            }
        }
    };
    auto write_a = [&](int buf) {
        for (int g = 0; g < 8; ++g) {
            int idx4 = tid + g * 128;
            int r = idx4 >> 3;
            int c = (idx4 & 7) << 2;
            *(float4*)&sAf[buf][r * LDA + c] = arg4[g];
        }
    };
#endif

    v8f acc[4][4];
    for (int m = 0; m < 4; ++m)
        for (int n = 0; n < 4; ++n)
            for (int i = 0; i < 8; ++i) acc[m][n][i] = 0.f;

    // ---- prologue: stage tile 0 into buffer 0 ----
#if ATHENA_TDM
    issue_a(0, 0);
    load_b_regs(0);
    write_b(0);
    wait_a();
#else
    load_a_regs(0);
    load_b_regs(0);
    write_a(0);
    write_b(0);
#endif
    __syncthreads();

    int cur = 0;
    for (int k0 = 0; k0 < K; k0 += BK) {
        const bool more = (k0 + BK) < K;
        if (more) {
            // start next tile's data movement (overlaps with compute below)
#if ATHENA_TDM
            issue_a(k0 + BK, cur ^ 1);
#else
            load_a_regs(k0 + BK);
#endif
            load_b_regs(k0 + BK);
        }

        // ---- compute on buffer `cur`: fragments + 16 WMMAs ----
        const float*  sa = &sAf[cur][0];
        const __bf16* sb = &sB[cur][0];
        v16bf af[4], bf[4];
        for (int m = 0; m < 4; ++m) {
            const float* p = sa + (size_t)(wm + m * 16 + lrow) * LDA;
            float4 q0 = *(const float4*)(p + kh);
            float4 q1 = *(const float4*)(p + kh + 4);
            float4 q2 = *(const float4*)(p + kh + 16);
            float4 q3 = *(const float4*)(p + kh + 20);
            af[m][0]  = (__bf16)q0.x; af[m][1]  = (__bf16)q0.y;
            af[m][2]  = (__bf16)q0.z; af[m][3]  = (__bf16)q0.w;
            af[m][4]  = (__bf16)q1.x; af[m][5]  = (__bf16)q1.y;
            af[m][6]  = (__bf16)q1.z; af[m][7]  = (__bf16)q1.w;
            af[m][8]  = (__bf16)q2.x; af[m][9]  = (__bf16)q2.y;
            af[m][10] = (__bf16)q2.z; af[m][11] = (__bf16)q2.w;
            af[m][12] = (__bf16)q3.x; af[m][13] = (__bf16)q3.y;
            af[m][14] = (__bf16)q3.z; af[m][15] = (__bf16)q3.w;
        }
        for (int n = 0; n < 4; ++n) {
            const __bf16* p = sb + (wn + n * 16 + lrow) * LDB;
            v8bf lo = *(const v8bf*)(p + kh);
            v8bf hi = *(const v8bf*)(p + kh + 16);
            for (int i = 0; i < 8; ++i) { bf[n][i] = lo[i]; bf[n][8 + i] = hi[i]; }
        }
        for (int m = 0; m < 4; ++m)
            for (int n = 0; n < 4; ++n)
                acc[m][n] = __builtin_amdgcn_wmma_f32_16x16x32_bf16(
                    false, af[m], false, bf[n], (short)0, acc[m][n], false, false);

        if (more) {
            // drain next tile into the alternate buffers, then hand off
#if ATHENA_TDM
            write_b(cur ^ 1);
            wait_a();
#else
            write_a(cur ^ 1);
            write_b(cur ^ 1);
#endif
            __syncthreads();
            cur ^= 1;
        }
    }

    // ---- epilogue; C/D layout: VGPR r -> row (r + 8*(lane>=16)), col = lane&15 ----
    const int rb = (lane >> 4) * 8;
    const int cl = lane & 15;
    for (int n = 0; n < 4; ++n) {
        int col = bn + wn + n * 16 + cl;
        if (!FAST && col >= N) continue;
        float bv = bias ? bias[col] : 0.f;
        for (int m = 0; m < 4; ++m) {
            int rowbase = bm + wm + m * 16 + rb;
            for (int r = 0; r < 8; ++r) {
                int row = rowbase + r;
                if (!FAST && row >= M) continue;
                float v = acc[m][n][r] + bv;
                if (mode == 1 || mode == 2) v = gelu_f(v);
                if (mode == 2 || mode == 3) v += R[(size_t)row * N + col];
                C[(size_t)row * N + col] = v;
            }
        }
    }
}

// ---------------- LayerNorm over H=256, one wave (32 lanes) per token ----------------
__global__ __launch_bounds__(256) void k_layernorm(
    const float* __restrict__ X, float* __restrict__ Y,
    const float* __restrict__ g, const float* __restrict__ b, int ntok)
{
    int wid = blockIdx.x * 8 + (threadIdx.x >> 5);
    int lane = threadIdx.x & 31;
    if (wid >= ntok) return;
    const float* row = X + (size_t)wid * HDIM;
    float v[8]; float s = 0.f;
    for (int i = 0; i < 8; ++i) { v[i] = row[lane + i * 32]; s += v[i]; }
    for (int o = 16; o > 0; o >>= 1) s += __shfl_xor(s, o, 32);
    float mean = s * (1.0f / HDIM);
    float var = 0.f;
    for (int i = 0; i < 8; ++i) { float d = v[i] - mean; var += d * d; }
    for (int o = 16; o > 0; o >>= 1) var += __shfl_xor(var, o, 32);
    float rstd = rsqrtf(var * (1.0f / HDIM) + 1e-5f);
    float* out = Y + (size_t)wid * HDIM;
    for (int i = 0; i < 8; ++i) {
        int c = lane + i * 32;
        out[c] = (v[i] - mean) * rstd * g[c] + b[c];
    }
}

// ---------------- softmax over head dim (32, contiguous): one wave per (token,head) ----------------
__global__ __launch_bounds__(256) void k_softmax32(float* __restrict__ X, int npairs)
{
    int wid = blockIdx.x * 8 + (threadIdx.x >> 5);
    int lane = threadIdx.x & 31;
    if (wid >= npairs) return;
    float* p = X + (size_t)wid * HD;
    float v = p[lane];
    float m = v;
    for (int o = 16; o > 0; o >>= 1) m = fmaxf(m, __shfl_xor(m, o, 32));
    float e = expf(v - m);
    float s = e;
    for (int o = 16; o > 0; o >>= 1) s += __shfl_xor(s, o, 32);
    p[lane] = e / s;
}

// ---- linear-attention context, T-split with atomic combine:
// ctx[d,e] += sum_t k[t,d] v[t,e]; ksum[d] += sum_t k[t,d]  over this block's T-chunk
__global__ __launch_bounds__(256) void k_ctx_part(
    const float* __restrict__ Km, const float* __restrict__ Vm,
    float* __restrict__ ctx, float* __restrict__ ksum, int T, int nsplit)
{
    __shared__ float sk[64][HD];
    __shared__ float sv[64][HD];
    int part = blockIdx.x % nsplit;
    int bh   = blockIdx.x / nsplit;
    int b = bh >> 3, h = bh & 7;
    int tlen = T / nsplit;          // multiple of 64 for all our shapes
    int tbeg = part * tlen;
    int tid = threadIdx.x;
    int d = tid >> 3;
    int e0 = (tid & 7) << 2;
    float a0 = 0.f, a1 = 0.f, a2 = 0.f, a3 = 0.f, ks = 0.f;
    const size_t baseo = (size_t)b * T * HDIM + h * HD;
    for (int t0 = tbeg; t0 < tbeg + tlen; t0 += 64) {
        __syncthreads();
        for (int g = 0; g < 8; ++g) {
            int idx = tid + g * 256;
            int tt = idx >> 5, dd = idx & 31;
            size_t go = baseo + (size_t)(t0 + tt) * HDIM + dd;
            sk[tt][dd] = Km[go];
            sv[tt][dd] = Vm[go];
        }
        __syncthreads();
        for (int t = 0; t < 64; ++t) {
            float kv = sk[t][d];
            a0 += kv * sv[t][e0 + 0];
            a1 += kv * sv[t][e0 + 1];
            a2 += kv * sv[t][e0 + 2];
            a3 += kv * sv[t][e0 + 3];
        }
        if (tid < HD)
            for (int t = 0; t < 64; ++t) ks += sk[t][tid];
    }
    float* cp = ctx + ((size_t)(b * NHEAD + h) * HD + d) * HD + e0;
    atomicAdd(cp + 0, a0);
    atomicAdd(cp + 1, a1);
    atomicAdd(cp + 2, a2);
    atomicAdd(cp + 3, a3);
    if (tid < HD) atomicAdd(&ksum[(size_t)(b * NHEAD + h) * HD + tid], ks);
}

// ---------------- apply: out = q + (q . ctx) / (q . ksum); wave per head, 32 tokens/block ----------------
__global__ __launch_bounds__(256) void k_attn_apply(
    const float* __restrict__ Q, const float* __restrict__ ctx,
    const float* __restrict__ ksum, float* __restrict__ Out, int T)
{
    __shared__ float sctx[NHEAD * HD * HD];   // 32 KB
    __shared__ float sks[NHEAD * HD];
    int b = blockIdx.x;
    int t0 = blockIdx.y * 32;
    int tid = threadIdx.x;
    for (int g = 0; g < 32; ++g)
        sctx[tid + g * 256] = ctx[(size_t)b * NHEAD * HD * HD + tid + g * 256];
    sks[tid] = ksum[(size_t)b * NHEAD * HD + tid];
    __syncthreads();
    int h = tid >> 5, lane = tid & 31;
    const float* sc = sctx + h * HD * HD;
    float ksl = sks[h * HD + lane];
    for (int tt = 0; tt < 32; ++tt) {
        size_t qi = ((size_t)b * T + (t0 + tt)) * HDIM + h * HD + lane;
        float q = Q[qi];
        float p = q * ksl;
        for (int o = 16; o > 0; o >>= 1) p += __shfl_xor(p, o, 32);
        float dinv = 1.0f / p;
        float num = 0.f;
        for (int dd = 0; dd < HD; ++dd) {
            float qd = __shfl(q, dd, 32);
            num += qd * sc[dd * HD + lane];
        }
        Out[qi] = q + num * dinv;
    }
}

// ---------------- host orchestration ----------------
// Leaf order (reference dict insertion order):
//  0:x 1:y0 | trunk@2 (pre.W,pre.b,mid0.W,mid0.b,mid1.W,mid1.b,post.W,post.b)
//  branch@10 (same 8) | blocks@18, stride 34:
//    +0..1 ln1.g/b, +2..3 ln2.g/b, +4..5 ln3, +6..7 ln4, +8..9 ln5,
//    +10..17 cross q.W/b k.W/b v.W/b proj.W/b, +18..25 self (same),
//    +26..29 mlp1 fc1.W/b fc2.W/b, +30..33 mlp2 | out@120 (8 leaves, post: 256x3)

extern "C" void kernel_launch(void* const* d_in, const int* in_sizes, int n_in,
                              void* d_out, int out_size, void* d_ws, size_t ws_size,
                              hipStream_t stream)
{
    (void)in_sizes; (void)n_in; (void)out_size; (void)ws_size;
    auto F = [&](int i) -> const float* { return (const float*)d_in[i]; };

    char* base = (char*)d_ws;
    size_t off = 0;
    auto alloc = [&](size_t bytes) -> char* {
        char* p = base + off;
        off += (bytes + 255) & ~(size_t)255;
        return p;
    };

    const size_t NT1 = (size_t)BATCH * T1;   // 65536 tokens
    const size_t NT2 = (size_t)BATCH * T2;   // 8192 tokens

    unsigned short* wbf = (unsigned short*)alloc((size_t)16 << 20);
    float* hbuf = (float*)alloc(NT1 * HDIM * 4);
    float* t0   = (float*)alloc(NT1 * HDIM * 4);
    float* t1   = (float*)alloc(NT1 * HDIM * 4);
    float* t2   = (float*)alloc(NT1 * HDIM * 4);
    float* t3   = (float*)alloc(NT1 * HDIM * 4);
    float* hid  = (float*)alloc((size_t)T1 * NINNER * 4);
    float* z0   = (float*)alloc(NT2 * HDIM * 4);
    float* zn   = (float*)alloc(NT2 * HDIM * 4);
    float* zk   = (float*)alloc(NT2 * HDIM * 4);
    float* zv   = (float*)alloc(NT2 * HDIM * 4);
    float* ctxb = (float*)alloc((size_t)BATCH * NHEAD * HD * HD * 4);
    float* ksum = (float*)alloc((size_t)BATCH * NHEAD * HD * 4);

    const int ctx_n  = BATCH * NHEAD * HD * HD;
    const int ksum_n = BATCH * NHEAD * HD;

    size_t woff = 0;
    auto cw = [&](int idx, int n) -> const unsigned short* {
        unsigned short* p = wbf + woff;
        woff += ((size_t)n + 63) & ~(size_t)63;
        k_cast_bf16<<<dim3((n + 255) / 256), dim3(256), 0, stream>>>(F(idx), p, n);
        return p;
    };
    auto gemm = [&](const float* A, const unsigned short* Wp, const float* b,
                    const float* Rr, float* Cc, int M, int N, int K, int mode) {
        dim3 g((M + 127) / 128, (N + 127) / 128);
        bool fast = ((K & 31) == 0) && ((N & 127) == 0) && ((M & 127) == 0);
        if (fast)
            k_gemm_wmma<true><<<g, dim3(128), 0, stream>>>(A, Wp, b, Rr, Cc, M, N, K, mode);
        else
            k_gemm_wmma<false><<<g, dim3(128), 0, stream>>>(A, Wp, b, Rr, Cc, M, N, K, mode);
    };
    auto lnorm = [&](const float* X, float* Y, int gi, int ntok) {
        k_layernorm<<<dim3((ntok + 7) / 8), dim3(256), 0, stream>>>(X, Y, F(gi), F(gi + 1), ntok);
    };
    auto smax = [&](float* X, int ntok) {
        int pairs = ntok * NHEAD;
        k_softmax32<<<dim3((pairs + 7) / 8), dim3(256), 0, stream>>>(X, pairs);
    };
    auto ctx_reduce = [&](const float* Kp, const float* Vp, int T, int nsplit) {
        k_zero<<<dim3((ctx_n + 255) / 256), dim3(256), 0, stream>>>(ctxb, ctx_n);
        k_zero<<<dim3((ksum_n + 255) / 256), dim3(256), 0, stream>>>(ksum, ksum_n);
        k_ctx_part<<<dim3(BATCH * NHEAD * nsplit), dim3(256), 0, stream>>>(
            Kp, Vp, ctxb, ksum, T, nsplit);
    };
    auto runmlp = [&](const float* X, int ntok, int Kin, int Nout, int bidx,
                      float* a, float* b2, float* c2, float* Y) {
        gemm(X, cw(bidx + 0, Kin * HDIM), F(bidx + 1), nullptr, a, ntok, HDIM, Kin, 1);
        gemm(a, cw(bidx + 2, HDIM * HDIM), F(bidx + 3), a, b2, ntok, HDIM, HDIM, 2);
        gemm(b2, cw(bidx + 4, HDIM * HDIM), F(bidx + 5), b2, c2, ntok, HDIM, HDIM, 2);
        gemm(c2, cw(bidx + 6, HDIM * Nout), F(bidx + 7), nullptr, Y, ntok, Nout, HDIM, 0);
    };

    // trunk / branch MLPs
    runmlp(F(0), (int)NT1, 3, HDIM, 2, t0, t1, t2, hbuf);
    runmlp(F(1), (int)NT2, 3, HDIM, 10, zn, zk, zv, z0);

    for (int blk = 0; blk < 3; ++blk) {
        int BB = 18 + blk * 34;
        // ---- cross attention ----
        lnorm(hbuf, t0, BB + 0, (int)NT1);
        lnorm(z0, zn, BB + 2, (int)NT2);
        gemm(t0, cw(BB + 10, HDIM * HDIM), F(BB + 11), nullptr, t1, (int)NT1, HDIM, HDIM, 0);
        smax(t1, (int)NT1);
        gemm(zn, cw(BB + 12, HDIM * HDIM), F(BB + 13), nullptr, zk, (int)NT2, HDIM, HDIM, 0);
        smax(zk, (int)NT2);
        gemm(zn, cw(BB + 14, HDIM * HDIM), F(BB + 15), nullptr, zv, (int)NT2, HDIM, HDIM, 0);
        ctx_reduce(zk, zv, T2, 2);
        k_attn_apply<<<dim3(BATCH, T1 / 32), dim3(256), 0, stream>>>(t1, ctxb, ksum, t2, T1);
        gemm(t2, cw(BB + 16, HDIM * HDIM), F(BB + 17), hbuf, hbuf, (int)NT1, HDIM, HDIM, 3);
        // ---- FFN 1 (chunk hidden per batch element) ----
        lnorm(hbuf, t0, BB + 4, (int)NT1);
        {
            const unsigned short* f1 = cw(BB + 26, HDIM * NINNER);
            const unsigned short* f2 = cw(BB + 28, NINNER * HDIM);
            for (int bb = 0; bb < BATCH; ++bb) {
                const float* Xc = t0 + (size_t)bb * T1 * HDIM;
                float* Hc = hbuf + (size_t)bb * T1 * HDIM;
                gemm(Xc, f1, F(BB + 27), nullptr, hid, T1, NINNER, HDIM, 1);
                gemm(hid, f2, F(BB + 29), Hc, Hc, T1, HDIM, NINNER, 3);
            }
        }
        // ---- self attention ----
        lnorm(hbuf, t0, BB + 6, (int)NT1);
        gemm(t0, cw(BB + 18, HDIM * HDIM), F(BB + 19), nullptr, t1, (int)NT1, HDIM, HDIM, 0);
        smax(t1, (int)NT1);
        gemm(t0, cw(BB + 20, HDIM * HDIM), F(BB + 21), nullptr, t2, (int)NT1, HDIM, HDIM, 0);
        smax(t2, (int)NT1);
        gemm(t0, cw(BB + 22, HDIM * HDIM), F(BB + 23), nullptr, t3, (int)NT1, HDIM, HDIM, 0);
        ctx_reduce(t2, t3, T1, 8);
        k_attn_apply<<<dim3(BATCH, T1 / 32), dim3(256), 0, stream>>>(t1, ctxb, ksum, t2, T1);
        gemm(t2, cw(BB + 24, HDIM * HDIM), F(BB + 25), hbuf, hbuf, (int)NT1, HDIM, HDIM, 3);
        // ---- FFN 2 ----
        lnorm(hbuf, t0, BB + 8, (int)NT1);
        {
            const unsigned short* f1 = cw(BB + 30, HDIM * NINNER);
            const unsigned short* f2 = cw(BB + 32, NINNER * HDIM);
            for (int bb = 0; bb < BATCH; ++bb) {
                const float* Xc = t0 + (size_t)bb * T1 * HDIM;
                float* Hc = hbuf + (size_t)bb * T1 * HDIM;
                gemm(Xc, f1, F(BB + 31), nullptr, hid, T1, NINNER, HDIM, 1);
                gemm(hid, f2, F(BB + 33), Hc, Hc, T1, HDIM, NINNER, 3);
            }
        }
    }

    // output MLP -> d_out [B*T1, 3] f32
    runmlp(hbuf, (int)NT1, HDIM, 3, 120, t0, t1, t2, (float*)d_out);
}
